// MultiHeadSelfAttention_44169443672070
// MI455X (gfx1250) — matrix operand order repaired and checked by reading, not compiled, not run on hardware
//
#include <hip/hip_runtime.h>
#include <hip/hip_bf16.h>
#include <stdint.h>

#define D_MODEL   1024
#define NUM_HEADS 16
#define HEAD_DIM  64
#define SEQ       2048
#define BATCH     2

typedef unsigned short u16;
typedef __attribute__((ext_vector_type(16))) __bf16 v16bf;
typedef __attribute__((ext_vector_type(8)))  float  v8f;
typedef __attribute__((ext_vector_type(4)))  unsigned int u32x4;
typedef int v4i __attribute__((vector_size(16)));

union Frag16 { v16bf v; u32x4 q[2]; u16 u[16]; };

#ifndef __has_builtin
#define __has_builtin(x) 0
#endif
#if __has_builtin(__builtin_amdgcn_global_load_async_to_lds_b128)
#define HAS_ASYNC_LDS 1
#else
#define HAS_ASYNC_LDS 0
#endif

#define AS1 __attribute__((address_space(1)))
#define AS3 __attribute__((address_space(3)))

typedef const AS1 u16   gc_u16;     // explicit global address space (forces global_load_*)
typedef const AS1 u32x4 gc_u32x4;

__device__ __forceinline__ u16 f2bf(float f) {
    uint32_t u = __float_as_uint(f);
    u += 0x7fffu + ((u >> 16) & 1u);          // round-to-nearest-even
    return (u16)(u >> 16);
}

// 16B global -> LDS copy: async ASYNCcnt path when available, sync fallback
__device__ __forceinline__ void async_cp16(const u16* g, u16* l) {
#if HAS_ASYNC_LDS
    __builtin_amdgcn_global_load_async_to_lds_b128((AS1 v4i*)g, (AS3 v4i*)l, 0, 0);
#else
    *(u32x4*)l = *(const u32x4*)g;
#endif
}
__device__ __forceinline__ void wait_async_0() {
#if HAS_ASYNC_LDS
#if __has_builtin(__builtin_amdgcn_s_wait_asynccnt)
    __builtin_amdgcn_s_wait_asynccnt(0);
#else
    asm volatile("s_wait_asynccnt 0x0" ::: "memory");
#endif
#endif
}
__device__ __forceinline__ void wait_async_3() {
#if HAS_ASYNC_LDS
#if __has_builtin(__builtin_amdgcn_s_wait_asynccnt)
    __builtin_amdgcn_s_wait_asynccnt(3);
#else
    asm volatile("s_wait_asynccnt 0x3" ::: "memory");
#endif
#endif
}

// ---------------------------------------------------------------------------
// fp32 -> bf16 conversion
// ---------------------------------------------------------------------------
__global__ void cvt_f32_bf16(const float* __restrict__ src, u16* __restrict__ dst, int n) {
    int i = blockIdx.x * blockDim.x + threadIdx.x;
    if (i < n) dst[i] = f2bf(src[i]);
}

// ---------------------------------------------------------------------------
// Tiled bf16 GEMM: C[M,N] = A[M,K] * W[N,K]^T, fp32 accumulate.
// Double-buffered LDS staging via async global->LDS (ASYNCcnt).
// MODE 0: fp32 row-major out; MODE 1: fused RoPE + bf16; MODE 2: bf16 [b,h,d,s].
// ---------------------------------------------------------------------------
#define GM 64
#define GN 128
#define GK 32
#define LPAD 40   // u16 row stride: 80B = 16B multiple

template<int MODE>
__global__ __launch_bounds__(256)
void gemm_bf16(const u16* __restrict__ A, const u16* __restrict__ W,
               void* __restrict__ out, int M, int N, int K) {
    __shared__ u16 As[2][GM][LPAD];
    __shared__ u16 Bs[2][GN][LPAD];

    const int tid   = threadIdx.x;
    const int lane  = tid & 31;
    const int l15   = lane & 15;
    const int lhalf = lane >> 4;
    const int wid   = tid >> 5;
    const int wm    = wid >> 2;     // 0..1
    const int wn    = wid & 3;      // 0..3
    const int m0    = blockIdx.y * GM;
    const int n0    = blockIdx.x * GN;

    v8f acc[2][2] = {};

    const int arow = tid >> 2;            // 0..63
    const int acol = (tid & 3) * 8;       // 0,8,16,24
    const int brow = tid >> 1;            // 0..127
    const int bcol = (tid & 1) * 16;      // 0,16

    const u16* agp = &A[(size_t)(m0 + arow) * K + acol];
    const u16* bgp = &W[(size_t)(n0 + brow) * K + bcol];

    // prologue: stage tile 0 into buffer 0
    async_cp16(agp,     &As[0][arow][acol]);
    async_cp16(bgp,     &Bs[0][brow][bcol]);
    async_cp16(bgp + 8, &Bs[0][brow][bcol + 8]);

    const int nk = K / GK;
    for (int t = 0; t < nk; t++) {
        const int buf = t & 1;
        if (t + 1 < nk) {
            agp += GK; bgp += GK;
            async_cp16(agp,     &As[buf ^ 1][arow][acol]);
            async_cp16(bgp,     &Bs[buf ^ 1][brow][bcol]);
            async_cp16(bgp + 8, &Bs[buf ^ 1][brow][bcol + 8]);
            wait_async_3();            // current tile landed; next tile in flight
        } else {
            wait_async_0();
        }
        __syncthreads();

        Frag16 a[2], b[2];
        #pragma unroll
        for (int i = 0; i < 2; i++) {
            const u16* ap = &As[buf][wm * 32 + i * 16 + l15][0];
            a[i].q[0] = *(const u32x4*)(ap + lhalf * 8);
            a[i].q[1] = *(const u32x4*)(ap + 16 + lhalf * 8);
        }
        #pragma unroll
        for (int j = 0; j < 2; j++) {
            const u16* bp = &Bs[buf][wn * 32 + j * 16 + l15][0];
            b[j].q[0] = *(const u32x4*)(bp + lhalf * 8);
            b[j].q[1] = *(const u32x4*)(bp + 16 + lhalf * 8);
        }
        #pragma unroll
        for (int i = 0; i < 2; i++)
            #pragma unroll
            for (int j = 0; j < 2; j++)
                acc[i][j] = __builtin_amdgcn_wmma_f32_16x16x32_bf16(
                    false, a[i].v, false, b[j].v, (short)0, acc[i][j], false, false);
        __syncthreads();
    }

    #pragma unroll
    for (int i = 0; i < 2; i++) {
        #pragma unroll
        for (int j = 0; j < 2; j++) {
            #pragma unroll
            for (int r = 0; r < 8; r++) {
                const int row = wm * 32 + i * 16 + ((lane < 16) ? r : r + 8);
                const int col = wn * 32 + j * 16 + l15;
                const int m = m0 + row;   // b*SEQ + s
                const int n = n0 + col;   // h*HEAD_DIM + dd
                float v = acc[i][j][r];
                if (MODE == 1) {
                    const int s  = m & (SEQ - 1);
                    const int dd = n & (HEAD_DIM - 1);
                    const float invf = __expf(-(float)(dd >> 1) * 0.28782313662f); // ln(1e4)/32
                    float sn, cs;
                    __sincosf((float)s * invf, &sn, &cs);
                    const float partner = __shfl_xor(v, 1, 32);
                    v = (lane & 1) ? (partner * sn + v * cs)
                                   : (v * cs - partner * sn);
                    ((u16*)out)[(size_t)m * N + n] = f2bf(v);
                } else if (MODE == 2) {
                    const int bb = m >> 11, s = m & (SEQ - 1);
                    const int h = n >> 6,  dd = n & (HEAD_DIM - 1);
                    ((u16*)out)[((size_t)(bb * NUM_HEADS + h) * HEAD_DIM + dd) * SEQ + s] = f2bf(v);
                } else {
                    ((float*)out)[(size_t)m * N + n] = v;
                }
            }
        }
    }
}

// ---------------------------------------------------------------------------
// Flash attention: 1 wave per (b, h, 16-query tile), 64-key tiles with online
// softmax. K/V row pointers are explicit addrspace(1) so fragment loads lower
// to global_load_b128 (LOADcnt only, clause-able) instead of flat_load.
// ---------------------------------------------------------------------------
#define PPAD 72   // u16 row stride: 144B = 16B multiple

__global__ __launch_bounds__(32)
void attn_fwd(const u16* __restrict__ Q, const u16* __restrict__ K,
              const u16* __restrict__ Vt, u16* __restrict__ ctx) {
    const int qt = blockIdx.x, h = blockIdx.y, b = blockIdx.z;
    const int lane  = threadIdx.x;
    const int l15   = lane & 15;
    const int lhalf = lane >> 4;
    __shared__ u16 Ps[16][PPAD];   // probability C-layout -> A-layout bounce

    Frag16 aQ[2];
    {
        const u16* qp = Q + (size_t)(b * SEQ + qt * 16 + l15) * D_MODEL + h * HEAD_DIM;
        #pragma unroll
        for (int d = 0; d < 2; d++) {
            aQ[d].q[0] = *(const u32x4*)(qp + d * 32 + lhalf * 16);
            aQ[d].q[1] = *(const u32x4*)(qp + d * 32 + lhalf * 16 + 8);
        }
    }

    // strength-reduced, explicitly-global row pointers (constant stride per tile)
    gc_u16* kptr[4];
    gc_u16* vptr[4];
    #pragma unroll
    for (int nh = 0; nh < 4; nh++) {
        kptr[nh] = (gc_u16*)(K + (size_t)(b * SEQ + nh * 16 + l15) * D_MODEL
                             + h * HEAD_DIM + lhalf * 16);
        vptr[nh] = (gc_u16*)(Vt + ((size_t)(b * NUM_HEADS + h) * HEAD_DIM + nh * 16 + l15) * SEQ
                             + lhalf * 16);
    }

    float m_run[8], l_run[8];
    v8f O[4] = {};
    #pragma unroll
    for (int r = 0; r < 8; r++) { m_run[r] = -1e30f; l_run[r] = 0.0f; }

    const int   rbase  = (lane < 16) ? 0 : 8;
    const int   q_hi   = qt * 16 + 15;
    const int   ntiles = (q_hi >> 6) + 1;     // 64-key tiles
    const float scale  = 0.125f;              // 1/sqrt(64)

    int kcol = l15;                           // this lane's key index (nh*16 added below)
    for (int kt = 0; kt < ntiles; kt++) {
        // ---- load all 8 K fragments up front (distinct registers) ----
        Frag16 bK[4][2];
        #pragma unroll
        for (int nh = 0; nh < 4; nh++) {
            #pragma unroll
            for (int d = 0; d < 2; d++) {
                bK[nh][d].q[0] = *(gc_u32x4*)(kptr[nh] + d * 32);
                bK[nh][d].q[1] = *(gc_u32x4*)(kptr[nh] + d * 32 + 8);
            }
        }
        // ---- 8 score WMMAs ----
        v8f c[4] = {};
        #pragma unroll
        for (int nh = 0; nh < 4; nh++)
            #pragma unroll
            for (int d = 0; d < 2; d++)
                c[nh] = __builtin_amdgcn_wmma_f32_16x16x32_bf16(
                    false, aQ[d].v, false, bK[nh][d].v, (short)0, c[nh], false, false);

        // ---- causal mask + scale + online softmax ----
        #pragma unroll
        for (int r = 0; r < 8; r++) {
            const int qrow = qt * 16 + rbase + r;
            float sc[4];
            #pragma unroll
            for (int nh = 0; nh < 4; nh++) {
                float s = c[nh][r] * scale;
                if (kcol + nh * 16 > qrow) s = -1e30f;
                sc[nh] = s;
            }
            float tm = fmaxf(fmaxf(sc[0], sc[1]), fmaxf(sc[2], sc[3]));
            tm = fmaxf(tm, __shfl_xor(tm, 1, 32));
            tm = fmaxf(tm, __shfl_xor(tm, 2, 32));
            tm = fmaxf(tm, __shfl_xor(tm, 4, 32));
            tm = fmaxf(tm, __shfl_xor(tm, 8, 32));
            const float mn    = fmaxf(m_run[r], tm);
            const float alpha = __expf(m_run[r] - mn);
            m_run[r] = mn;
            float e[4], rs = 0.0f;
            #pragma unroll
            for (int nh = 0; nh < 4; nh++) { e[nh] = __expf(sc[nh] - mn); rs += e[nh]; }
            rs += __shfl_xor(rs, 1, 32);
            rs += __shfl_xor(rs, 2, 32);
            rs += __shfl_xor(rs, 4, 32);
            rs += __shfl_xor(rs, 8, 32);
            l_run[r] = l_run[r] * alpha + rs;
            #pragma unroll
            for (int nh = 0; nh < 4; nh++) O[nh][r] *= alpha;
            #pragma unroll
            for (int nh = 0; nh < 4; nh++)
                Ps[rbase + r][nh * 16 + l15] = f2bf(e[nh]);
        }
        __syncthreads();

        // ---- P fragments (A-layout), 2 chunks of 32 keys ----
        Frag16 aP[2];
        #pragma unroll
        for (int ck = 0; ck < 2; ck++) {
            const u16* pp = &Ps[l15][ck * 32];
            aP[ck].q[0] = *(const u32x4*)(pp + lhalf * 16);
            aP[ck].q[1] = *(const u32x4*)(pp + lhalf * 16 + 8);
        }

        // ---- load all 8 V fragments up front, then 8 PV WMMAs ----
        Frag16 bV[4][2];
        #pragma unroll
        for (int nh = 0; nh < 4; nh++) {
            #pragma unroll
            for (int ck = 0; ck < 2; ck++) {
                bV[nh][ck].q[0] = *(gc_u32x4*)(vptr[nh] + ck * 32);
                bV[nh][ck].q[1] = *(gc_u32x4*)(vptr[nh] + ck * 32 + 8);
            }
        }
        #pragma unroll
        for (int nh = 0; nh < 4; nh++)
            #pragma unroll
            for (int ck = 0; ck < 2; ck++)
                O[nh] = __builtin_amdgcn_wmma_f32_16x16x32_bf16(
                    false, aP[ck].v, false, bV[nh][ck].v, (short)0, O[nh], false, false);
        __syncthreads();

        // advance by one 64-key tile
        kcol += 64;
        #pragma unroll
        for (int nh = 0; nh < 4; nh++) {
            kptr[nh] += 64 * D_MODEL;
            vptr[nh] += 64;
        }
    }

    #pragma unroll
    for (int nh = 0; nh < 4; nh++) {
        #pragma unroll
        for (int r = 0; r < 8; r++) {
            const float o = O[nh][r] / l_run[r];
            ctx[(size_t)(b * SEQ + qt * 16 + rbase + r) * D_MODEL
                + h * HEAD_DIM + nh * 16 + l15] = f2bf(o);
        }
    }
}

// ---------------------------------------------------------------------------
extern "C" void kernel_launch(void* const* d_in, const int* in_sizes, int n_in,
                              void* d_out, int out_size, void* d_ws, size_t ws_size,
                              hipStream_t stream) {
    const float* x  = (const float*)d_in[0];
    const float* Wq = (const float*)d_in[1];
    const float* Wk = (const float*)d_in[2];
    const float* Wv = (const float*)d_in[3];
    const float* Wo = (const float*)d_in[4];

    const int XE = BATCH * SEQ * D_MODEL;   // 4,194,304
    const int WE = D_MODEL * D_MODEL;       // 1,048,576

    u16* xbf = (u16*)d_ws;
    u16* wq  = xbf + XE;
    u16* wk  = wq + WE;
    u16* wv  = wk + WE;
    u16* wo  = wv + WE;
    u16* Qb  = wo + WE;
    u16* Kb  = Qb + XE;
    u16* Vt  = Kb + XE;
    u16* ctx = Vt + XE;
    // total ws use: (5*XE + 4*WE) * 2 bytes = 48 MiB

    cvt_f32_bf16<<<XE / 256, 256, 0, stream>>>(x,  xbf, XE);
    cvt_f32_bf16<<<WE / 256, 256, 0, stream>>>(Wq, wq,  WE);
    cvt_f32_bf16<<<WE / 256, 256, 0, stream>>>(Wk, wk,  WE);
    cvt_f32_bf16<<<WE / 256, 256, 0, stream>>>(Wv, wv,  WE);
    cvt_f32_bf16<<<WE / 256, 256, 0, stream>>>(Wo, wo,  WE);

    dim3 ggrid(D_MODEL / GN, (BATCH * SEQ) / GM);
    gemm_bf16<1><<<ggrid, 256, 0, stream>>>(xbf, wq, Qb, BATCH * SEQ, D_MODEL, D_MODEL);
    gemm_bf16<1><<<ggrid, 256, 0, stream>>>(xbf, wk, Kb, BATCH * SEQ, D_MODEL, D_MODEL);
    gemm_bf16<2><<<ggrid, 256, 0, stream>>>(xbf, wv, Vt, BATCH * SEQ, D_MODEL, D_MODEL);

    attn_fwd<<<dim3(SEQ / 16, NUM_HEADS, BATCH), 32, 0, stream>>>(Qb, Kb, Vt, ctx);

    gemm_bf16<0><<<ggrid, 256, 0, stream>>>(ctx, wo, (float*)d_out,
                                            BATCH * SEQ, D_MODEL, D_MODEL);
}
